// DecoderBlock_51127290692115
// MI455X (gfx1250) — matrix operand order repaired and checked by reading, not compiled, not run on hardware
//
#include <hip/hip_runtime.h>
#include <hip/hip_bf16.h>
#include <stdint.h>

// ---------------------------------------------------------------------------
// DecoderBlock: GNN edge block on gfx1250 (wave32, WMMA bf16 -> f32 acc).
//   z:         [50000, 64]  f32
//   edge_attr: [800000, 32] f32
//   edge_index:[2, 800000]  i32   (row 0 = src, row 1 = dst)
//   z_edge = relu(cat(z_src, z_dst, ea) @ W_ffw + b_ffw)          [E,160]
//   msg    = sigmoid(m@W_f+b_f) * softplus(m@W_s+b_s), m=cat(z_dst,z_src)
//   z_node = z + segment_sum(msg, dst)                            [N,64]
// Output: d_out = [ z_node (50000*64) | z_edge (800000*160) ] f32
// ---------------------------------------------------------------------------

typedef __attribute__((ext_vector_type(16))) __bf16 v16bf;
typedef __attribute__((ext_vector_type(8)))  float  v8f;

#define N_NODES  50000
#define N_EDGES  800000
#define HIDDEN   64
#define EDGE_F   32
#define CATD     160   // 2H + F
#define M2D      128   // 2H
#define WPB      4     // waves per block

union FragB16 { v16bf v; uint4 q[2]; };
union AccU    { v8f   v; float f[8]; };

__device__ __forceinline__ uint32_t f32_to_bf16_rne(float x) {
  union { float f; uint32_t u; } v; v.f = x;
  uint32_t r = v.u + 0x7FFFu + ((v.u >> 16) & 1u);
  return r >> 16;
}
__device__ __forceinline__ uint32_t pk_bf16(float a, float b) {
  return f32_to_bf16_rne(a) | (f32_to_bf16_rne(b) << 16);
}

// ---- weight prep: Wt[n*K + k] = bf16(W[k*N + n])  (transpose + convert) ----
__global__ void wprep_kernel(const float* __restrict__ W,
                             uint16_t* __restrict__ Wt, int K, int N) {
  int tid = blockIdx.x * blockDim.x + threadIdx.x;
  if (tid >= K * N) return;
  int n = tid / K, k = tid - n * K;
  Wt[tid] = (uint16_t)f32_to_bf16_rne(W[(size_t)k * N + n]);
}

// ---- z_node init: residual base (d_out is poisoned; rewrite every call) ----
__global__ void zcopy_kernel(const float* __restrict__ z, float* __restrict__ out) {
  int tid = blockIdx.x * blockDim.x + threadIdx.x;
  if (tid < N_NODES * HIDDEN) out[tid] = z[tid];
}

// B fragment (32x16 bf16, contiguous k order, same as A): lane = column (lane&15),
// lanes 0-15 hold K 0..15, lanes 16-31 hold K 16..31 of the k-tile.
// Weights are stored N-major (transposed), so this is 32 contiguous bytes.
__device__ __forceinline__ v16bf load_bfrag(const uint16_t* __restrict__ Wt,
                                            int Kdim, int nt, int kt, int lane) {
  int n   = nt * 16 + (lane & 15);
  int off = n * Kdim + kt * 32 + ((lane >= 16) ? 16 : 0);
  FragB16 b;
  b.q[0] = *(const uint4*)(Wt + off);
  b.q[1] = *(const uint4*)(Wt + off + 8);
  return b.v;
}

__global__ __launch_bounds__(32 * WPB) void edge_kernel(
    const float* __restrict__ z, const float* __restrict__ ea,
    const int* __restrict__ eidx,
    const float* __restrict__ b_ffw, const float* __restrict__ b_f,
    const float* __restrict__ b_s,
    const uint16_t* __restrict__ Wffw_t, const uint16_t* __restrict__ Wf_t,
    const uint16_t* __restrict__ Ws_t,
    float* __restrict__ z_node, float* __restrict__ z_edge) {
  // Per-wave private staging: 16x160 bf16 A-tile + 16 dst indices.
  __shared__ __align__(16) uint16_t tileLds[WPB][16][CATD];
  __shared__ int dstLds[WPB][16];

  const int lane = threadIdx.x & 31;
  const int w    = threadIdx.x >> 5;
  const int tile = blockIdx.x * WPB + w;
  const int edgeBase = tile * 16;         // E = 800000 = 50000 * 16, exact
  const int r    = lane & 15;             // edge row within tile
  const bool hi  = lane >= 16;
  const int e    = edgeBase + r;

  // ---------------- gather + f32->bf16 convert into LDS ----------------
  if (!hi) {
    const int srcIdx = eidx[e];
    const float4* zr = (const float4*)(z + (size_t)srcIdx * HIDDEN);
#pragma unroll
    for (int i = 0; i < 16; ++i) {        // z_src -> cols [0,64)
      float4 t = zr[i];
      *(uint2*)&tileLds[w][r][i * 4] = make_uint2(pk_bf16(t.x, t.y), pk_bf16(t.z, t.w));
    }
    const float4* er = (const float4*)(ea + (size_t)e * EDGE_F);
#pragma unroll
    for (int i = 0; i < 4; ++i) {         // ea[0:16) -> cols [128,144)
      float4 t = er[i];
      *(uint2*)&tileLds[w][r][128 + i * 4] = make_uint2(pk_bf16(t.x, t.y), pk_bf16(t.z, t.w));
    }
  } else {
    const int dstIdx = eidx[N_EDGES + e];
    dstLds[w][r] = dstIdx;
    const float4* zr = (const float4*)(z + (size_t)dstIdx * HIDDEN);
#pragma unroll
    for (int i = 0; i < 16; ++i) {        // z_dst -> cols [64,128)
      float4 t = zr[i];
      *(uint2*)&tileLds[w][r][64 + i * 4] = make_uint2(pk_bf16(t.x, t.y), pk_bf16(t.z, t.w));
    }
    const float4* er = (const float4*)(ea + (size_t)e * EDGE_F);
#pragma unroll
    for (int i = 4; i < 8; ++i) {         // ea[16:32) -> cols [144,160)
      float4 t = er[i];
      *(uint2*)&tileLds[w][r][128 + i * 4] = make_uint2(pk_bf16(t.x, t.y), pk_bf16(t.z, t.w));
    }
  }
  // LDS is wave-private; compiler's DScnt waits order the in-wave RAW.

  // ---- A fragments (16x32 bf16 layout): lane row = lane&15,
  //      lanes 0-15: K 0..15; lanes 16-31: K 16..31 ----
  FragB16 a[5];
#pragma unroll
  for (int f = 0; f < 5; ++f) {
    const int kb = f * 32;
    a[f].q[0] = *(const uint4*)&tileLds[w][r][kb + (hi ? 16 : 0)];
    a[f].q[1] = *(const uint4*)&tileLds[w][r][kb + (hi ? 24 : 8)];
  }
  // Fragment meaning: a0,a1 = z_src K0-63; a2,a3 = z_dst; a4 = ea.
  // CGConv A = cat(z_dst, z_src) = fragments {2,3,0,1}.

  // ---------------- GEMM1: z_edge = relu(A160 @ W_ffw + b) ----------------
  const int col0 = lane & 15;
  const int rb   = hi ? 8 : 0;            // C layout: M = vgpr + (hi ? 8 : 0)
#pragma unroll
  for (int nt = 0; nt < 10; ++nt) {
    const int col = nt * 16 + col0;
    const float bias = b_ffw[col];
    AccU acc;
#pragma unroll
    for (int i = 0; i < 8; ++i) acc.f[i] = bias;
#pragma unroll
    for (int f = 0; f < 5; ++f) {
      v16bf b = load_bfrag(Wffw_t, CATD, nt, f, lane);
      acc.v = __builtin_amdgcn_wmma_f32_16x16x32_bf16(
          false, a[f].v, false, b, (short)0, acc.v, false, false);
    }
#pragma unroll
    for (int vv = 0; vv < 8; ++vv) {
      float x = acc.f[vv];
      z_edge[(size_t)(edgeBase + rb + vv) * CATD + col] = x > 0.f ? x : 0.f;
    }
  }

  // ------- GEMM2/3: gate/core, then sigmoid*softplus, scatter-add -------
  const int perm[4] = {2, 3, 0, 1};
#pragma unroll
  for (int nt = 0; nt < 4; ++nt) {
    const int col = nt * 16 + col0;       // hidden channel h
    const float bf = b_f[col], bs = b_s[col];
    AccU accF, accS;
#pragma unroll
    for (int i = 0; i < 8; ++i) { accF.f[i] = bf; accS.f[i] = bs; }
#pragma unroll
    for (int f = 0; f < 4; ++f) {
      v16bf av = a[perm[f]].v;
      v16bf bF = load_bfrag(Wf_t, M2D, nt, f, lane);
      accF.v = __builtin_amdgcn_wmma_f32_16x16x32_bf16(
          false, av, false, bF, (short)0, accF.v, false, false);
      v16bf bS = load_bfrag(Ws_t, M2D, nt, f, lane);
      accS.v = __builtin_amdgcn_wmma_f32_16x16x32_bf16(
          false, av, false, bS, (short)0, accS.v, false, false);
    }
#pragma unroll
    for (int vv = 0; vv < 8; ++vv) {
      const float g  = 1.f / (1.f + __expf(-accF.f[vv]));
      const float s  = accS.f[vv];
      const float sp = (s > 20.f) ? s : log1pf(__expf(s));
      const int   d  = dstLds[w][rb + vv];
      atomicAdd(&z_node[(size_t)d * HIDDEN + col], g * sp);
    }
  }
}

extern "C" void kernel_launch(void* const* d_in, const int* in_sizes, int n_in,
                              void* d_out, int out_size, void* d_ws, size_t ws_size,
                              hipStream_t stream) {
  const float* z        = (const float*)d_in[0];
  const float* edge_attr= (const float*)d_in[1];
  const int*   edge_idx = (const int*)d_in[2];
  const float* W_ffw    = (const float*)d_in[3];
  const float* b_ffw    = (const float*)d_in[4];
  const float* W_f      = (const float*)d_in[5];
  const float* b_f      = (const float*)d_in[6];
  const float* W_s      = (const float*)d_in[7];
  const float* b_s      = (const float*)d_in[8];

  float* z_node = (float*)d_out;                               // [N*H]
  float* z_edge = (float*)d_out + (size_t)N_NODES * HIDDEN;    // [E*160]

  // Workspace: bf16 transposed weights (N-major) -> B-fragments are
  // contiguous 32B per lane. Total 83,968 bytes.
  uint16_t* Wffw_t = (uint16_t*)d_ws;                 // 160*160
  uint16_t* Wf_t   = Wffw_t + CATD * CATD;            // 64*128
  uint16_t* Ws_t   = Wf_t + HIDDEN * M2D;             // 64*128

  wprep_kernel<<<(CATD * CATD + 255) / 256, 256, 0, stream>>>(W_ffw, Wffw_t, CATD, CATD);
  wprep_kernel<<<(M2D * HIDDEN + 255) / 256, 256, 0, stream>>>(W_f, Wf_t, M2D, HIDDEN);
  wprep_kernel<<<(M2D * HIDDEN + 255) / 256, 256, 0, stream>>>(W_s, Ws_t, M2D, HIDDEN);
  zcopy_kernel<<<(N_NODES * HIDDEN + 255) / 256, 256, 0, stream>>>(z, z_node);

  const int tiles  = N_EDGES / 16;     // 50000
  const int blocks = tiles / WPB;      // 12500
  edge_kernel<<<blocks, 32 * WPB, 0, stream>>>(
      z, edge_attr, edge_idx, b_ffw, b_f, b_s, Wffw_t, Wf_t, Ws_t,
      z_node, z_edge);
}
